// GPTBlock_1056561955091
// MI455X (gfx1250) — matrix operand order repaired and checked by reading, compiled, not run on hardware
//
// GPT block forward for MI455X (gfx1250, wave32, WMMA + TDM).
// ~138 GFLOP vs ~60MB traffic -> hard compute-bound, so every matmul uses
// v_wmma_f32_16x16x32_bf16 (bf16 in, fp32 acc) — the high-rate matrix path
// on CDNA5 for float data (fp32 WMMA is K=4 only). GEMM A-tiles are staged
// into LDS by the Tensor Data Mover (tensor_load_to_lds + s_wait_tensorcnt);
// B-tiles are transposed per-thread (TDM cannot transpose) with a
// global_prefetch of the next tile.
#include <hip/hip_runtime.h>
#include <cstdint>

#define EMBED 1024
#define HEADS 16
#define HDIM  64
#define FFDIM 4096
#define BATCH 2
#define SEQ   2048
#define MROWS (BATCH*SEQ)   // 4096

typedef unsigned short u16;
typedef unsigned int   u32;
typedef __attribute__((ext_vector_type(16))) __bf16 v16bf;
typedef __attribute__((ext_vector_type(8)))  float  v8f;
typedef __attribute__((ext_vector_type(4)))  unsigned int v4u;
typedef __attribute__((ext_vector_type(8)))  int    v8i;
typedef __attribute__((ext_vector_type(4)))  int    v4i;

union Frag { v16bf v; uint4 q[2]; };

// round-to-nearest-even f32 -> bf16
__device__ inline u16 f2b(float x) {
  u32 u = __float_as_uint(x);
  u32 r = (u + 0x7fffu + ((u >> 16) & 1u)) >> 16;
  return (u16)r;
}

__device__ inline v8f wmma_bf16(const Frag& a, const Frag& b, v8f c) {
  // (neg_a, A, neg_b, B, c_mod, C, reuse_a, reuse_b)
  return __builtin_amdgcn_wmma_f32_16x16x32_bf16(false, a.v, false, b.v,
                                                 (short)0, c, false, false);
}

__device__ inline float gelu_f(float v) {
  return 0.5f * v * (1.0f + erff(v * 0.70710678118654752f));
}

// ---------------------------------------------------------------------------
// elementwise f32 -> bf16 (weights)
__global__ void cvt_bf16_k(const float* __restrict__ in, u16* __restrict__ out, int n) {
  int i = blockIdx.x * blockDim.x + threadIdx.x;
  if (i < n) out[i] = f2b(in[i]);
}

// ---------------------------------------------------------------------------
// LayerNorm over rows of EMBED fp32 -> bf16. One 256-thread block per row.
__global__ void __launch_bounds__(256) layernorm_bf16_k(
    const float* __restrict__ X, const float* __restrict__ gamma,
    const float* __restrict__ beta, u16* __restrict__ Y) {
  int row = blockIdx.x;
  int t = threadIdx.x;
  const float* xr = X + (size_t)row * EMBED;
  float4 v = ((const float4*)xr)[t];
  float s  = v.x + v.y + v.z + v.w;
  float ss = v.x*v.x + v.y*v.y + v.z*v.z + v.w*v.w;
  __shared__ float rs[256], rq[256];
  rs[t] = s; rq[t] = ss;
  __syncthreads();
  for (int o = 128; o > 0; o >>= 1) {
    if (t < o) { rs[t] += rs[t + o]; rq[t] += rq[t + o]; }
    __syncthreads();
  }
  float mu   = rs[0] * (1.0f / EMBED);
  float var  = rq[0] * (1.0f / EMBED) - mu * mu;
  float rstd = rsqrtf(var + 1e-5f);
  int c0 = t * 4;
  u16* yr = Y + (size_t)row * EMBED + c0;
  float vv[4] = { v.x, v.y, v.z, v.w };
#pragma unroll
  for (int j = 0; j < 4; j++)
    yr[j] = f2b((vv[j] - mu) * rstd * gamma[c0 + j] + beta[c0 + j]);
}

// ---------------------------------------------------------------------------
// Generic bf16 WMMA GEMM: C[M,N] = A[M,K] @ B[K,N] (+bias)(gelu)(+resid)
// 128x128 block tile, 8 waves, each wave = 2x4 grid of 16x16 WMMA tiles.
// A-tile staged by TDM (tensor_load_to_lds), B-tile transposed per-thread.
enum { EP_GELU = 1, EP_RES = 2, EP_F32 = 4, EP_B16 = 8, EP_VT = 16, EP_BIAS = 32 };

template <int FLAGS>
__global__ void __launch_bounds__(256) gemm_bf16_k(
    const u16* __restrict__ A, const u16* __restrict__ Bw,
    const float* __restrict__ bias, const float* __restrict__ resid,
    float* __restrict__ outF, u16* __restrict__ outB,
    int M, int N, int K) {
  __shared__ alignas(16) u16 sA[128 * 32];   // [r][k], row-major, stride 32
  __shared__ alignas(16) u16 sB[128 * 32];   // [n][k], transposed, stride 32
  int t = threadIdx.x;
  int lane = t & 31, w = t >> 5;
  int nIdx = lane & 15, g = lane >> 4;
  int m0 = blockIdx.y * 128, n0 = blockIdx.x * 128;
  int wm = (w & 3) * 32, wn = (w >> 2) * 64;

  v8f zf = {};
  v8f acc[2][4];
#pragma unroll
  for (int a2 = 0; a2 < 2; a2++)
#pragma unroll
    for (int b2 = 0; b2 < 4; b2++) acc[a2][b2] = zf;

  // LDS byte offset of sA: generic LDS pointers carry the offset in addr[31:0]
  unsigned ldsA = (unsigned)(uintptr_t)(&sA[0]);

  for (int kc = 0; kc < K; kc += 32) {
    __syncthreads();   // previous iteration's consumers done
    if (t < 32) {
      // --- TDM: DMA the 128x32-half A tile into LDS (D# per ISA 08 §8.3/8.4)
      unsigned long long ga =
          (unsigned long long)(uintptr_t)(A + (size_t)m0 * K + kc);
      unsigned rows = (unsigned)(M - m0);          // tensor rows from tile start
      v4u g0;
      g0.x = 1u;                                   // count=1, no gather
      g0.y = ldsA;                                 // lds_addr (bytes)
      g0.z = (unsigned)ga;                         // global_addr[31:0]
      g0.w = (unsigned)((ga >> 32) & 0x01FFFFFFu)  // global_addr[56:32]
             | 0x80000000u;                        // type=2 ("image"), bits 127:126 = 10b
      v8i g1;
      g1[0] = 0x00010000;                          // data_size=1 (2 bytes/elem)
      g1[1] = (int)(((unsigned)K & 0xFFFFu) << 16);            // tensor_dim0 lo
      g1[2] = (int)((((unsigned)K >> 16) & 0xFFFFu)
                    | ((rows & 0xFFFFu) << 16));               // dim0 hi | dim1 lo
      g1[3] = (int)(((rows >> 16) & 0xFFFFu) | (32u << 16));   // dim1 hi | tile_dim0=32
      g1[4] = 128;                                 // tile_dim1=128, tile_dim2=0
      g1[5] = K;                                   // tensor_dim0_stride lo (elems)
      g1[6] = 0;                                   // stride hi | dim1_stride lo
      g1[7] = 0;
      v4i gz4 = { 0, 0, 0, 0 };                    // 2D tensor: groups 2/3 unused
      v8i gz8 = { 0, 0, 0, 0, 0, 0, 0, 0 };
      __builtin_amdgcn_tensor_load_to_lds(g0, g1, gz4, gz4, gz8, 0);
    }
    {   // stage B tile transposed: read 16 contiguous n, scatter to [n][k]
      int kk = t >> 3, nn = (t & 7) * 16;
      const u16* src = Bw + (size_t)(kc + kk) * N + n0 + nn;
      if (kc + 32 < K) __builtin_prefetch(src + (size_t)32 * N, 0, 1);
      uint4 b0 = *(const uint4*)(src);
      uint4 b1 = *(const uint4*)(src + 8);
      alignas(16) u16 tmp[16];
      *(uint4*)(tmp)     = b0;
      *(uint4*)(tmp + 8) = b1;
#pragma unroll
      for (int i2 = 0; i2 < 16; i2++) sB[(nn + i2) * 32 + kk] = tmp[i2];
    }
    if (t < 32) __builtin_amdgcn_s_wait_tensorcnt(0);   // TDM landed in LDS
    __syncthreads();
    // A fragments (16-bit A layout: lane m=lane&15, k chunks {8g..} and {16+8g..})
    Frag aF0, aF1;
    const u16* ap0 = sA + (wm + nIdx) * 32 + 8 * g;
    aF0.q[0] = *(const uint4*)(ap0);
    aF0.q[1] = *(const uint4*)(ap0 + 16);
    const u16* ap1 = sA + (wm + 16 + nIdx) * 32 + 8 * g;
    aF1.q[0] = *(const uint4*)(ap1);
    aF1.q[1] = *(const uint4*)(ap1 + 16);
#pragma unroll
    for (int nt = 0; nt < 4; nt++) {
      // B fragment: lane n=lane&15, k chunks {8g..} and {16+8g..}
      const u16* bp = sB + (wn + nt * 16 + nIdx) * 32 + 8 * g;
      Frag bF;
      bF.q[0] = *(const uint4*)(bp);
      bF.q[1] = *(const uint4*)(bp + 16);
      acc[0][nt] = wmma_bf16(aF0, bF, acc[0][nt]);
      acc[1][nt] = wmma_bf16(aF1, bF, acc[1][nt]);
    }
  }

  // epilogue (compile-time specialized): C/D layout lane->n, vgpr i -> m=i+8g
#pragma unroll
  for (int mt = 0; mt < 2; mt++)
#pragma unroll
    for (int nt = 0; nt < 4; nt++)
#pragma unroll
      for (int i = 0; i < 8; i++) {
        int m = m0 + wm + mt * 16 + i + 8 * g;
        int n = n0 + wn + nt * 16 + nIdx;
        float vv = acc[mt][nt][i];
        if (FLAGS & EP_BIAS) vv += bias[n];
        if (FLAGS & EP_GELU) vv = gelu_f(vv);
        if (FLAGS & EP_RES)  vv += resid[(size_t)m * N + n];
        if (FLAGS & EP_F32)  outF[(size_t)m * N + n] = vv;
        if (FLAGS & EP_B16)  outB[(size_t)m * N + n] = f2b(vv);
        if (FLAGS & EP_VT) {           // V transposed to [b][h][d][s] for PV step
          int bb = m >> 11, s = m & (SEQ - 1);
          int hh = n >> 6,  d = n & (HDIM - 1);
          outB[((size_t)((bb * HEADS + hh) * HDIM + d)) * SEQ + s] = f2b(vv);
        }
      }
}

// ---------------------------------------------------------------------------
// Flash attention: one wave per (b, h, 16-query tile). Causal, scale 1/32.
// Q,K in [b*S+s][E] bf16 (head at col h*64), V transposed [b][h][d][s] bf16.
__global__ void __launch_bounds__(32) flash_attn_k(
    const u16* __restrict__ Q, const u16* __restrict__ Kb,
    const u16* __restrict__ VT, u16* __restrict__ Z) {
  __shared__ alignas(16) u16 sP[16 * 32];
  int lane = threadIdx.x;
  int nIdx = lane & 15, g = lane >> 4;
  int q0 = blockIdx.x * 16;
  int h  = blockIdx.y;
  int b  = blockIdx.z;
  int bS = b * SEQ;

  // Q as A fragments for d-chunks [0,32) and [32,64)
  Frag aQ0, aQ1;
  {
    const u16* qrow = Q + ((size_t)(bS + q0 + nIdx)) * EMBED + h * HDIM;
    aQ0.q[0] = *(const uint4*)(qrow + 8 * g);
    aQ0.q[1] = *(const uint4*)(qrow + 8 * g + 16);
    aQ1.q[0] = *(const uint4*)(qrow + 32 + 8 * g);
    aQ1.q[1] = *(const uint4*)(qrow + 32 + 8 * g + 16);
  }
  v8f zf = {};
  v8f o0 = zf, o1 = zf, o2 = zf, o3 = zf;
  float mrow[8], srow[8];
#pragma unroll
  for (int i = 0; i < 8; i++) { mrow[i] = -__builtin_inff(); srow[i] = 0.f; }

  const u16* vbase = VT + ((size_t)((b * HEADS + h) * HDIM)) * SEQ;

  for (int j0 = 0; j0 < q0 + 16; j0 += 32) {   // 32 keys per chunk
    v8f ct[2];
#pragma unroll
    for (int tt = 0; tt < 2; tt++) {
      // K^T as B operand: lane n = key, contraction d chunks {8g..} and {16+8g..}
      const u16* krow = Kb + ((size_t)(bS + j0 + 16 * tt + nIdx)) * EMBED + h * HDIM;
      Frag bk0, bk1;
      bk0.q[0] = *(const uint4*)(krow + 8 * g);
      bk0.q[1] = *(const uint4*)(krow + 8 * g + 16);
      bk1.q[0] = *(const uint4*)(krow + 32 + 8 * g);
      bk1.q[1] = *(const uint4*)(krow + 32 + 8 * g + 16);
      v8f c = zf;
      c = wmma_bf16(aQ0, bk0, c);
      c = wmma_bf16(aQ1, bk1, c);
      ct[tt] = c;
    }
    // online softmax: row m = i + 8g lives on one 16-lane group -> shfl_xor reduce
    float p0[8], p1[8];
#pragma unroll
    for (int i = 0; i < 8; i++) {
      int qi = q0 + i + 8 * g;
      float v0 = ct[0][i] * 0.03125f;          // 1/sqrt(EMBED) = 1/32
      float v1 = ct[1][i] * 0.03125f;
      if (j0 + nIdx      > qi) v0 = -__builtin_inff();
      if (j0 + 16 + nIdx > qi) v1 = -__builtin_inff();
      float mx = fmaxf(v0, v1);
#pragma unroll
      for (int sft = 8; sft >= 1; sft >>= 1) mx = fmaxf(mx, __shfl_xor(mx, sft, 32));
      float nm  = fmaxf(mrow[i], mx);
      float fac = __expf(mrow[i] - nm);
      float e0 = __expf(v0 - nm);
      float e1 = __expf(v1 - nm);
      float sm = e0 + e1;
#pragma unroll
      for (int sft = 8; sft >= 1; sft >>= 1) sm += __shfl_xor(sm, sft, 32);
      srow[i] = srow[i] * fac + sm;
      mrow[i] = nm;
      o0[i] *= fac; o1[i] *= fac; o2[i] *= fac; o3[i] *= fac;
      p0[i] = e0; p1[i] = e1;
    }
    // re-shape P from C-layout (lane->key) to A-layout (lane->query) via LDS
#pragma unroll
    for (int i = 0; i < 8; i++) {
      sP[(i + 8 * g) * 32 + nIdx]      = f2b(p0[i]);
      sP[(i + 8 * g) * 32 + 16 + nIdx] = f2b(p1[i]);
    }
    asm volatile("s_wait_dscnt 0" ::: "memory");
    Frag aP;
    {
      const u16* prow = sP + nIdx * 32 + 16 * g;
      aP.q[0] = *(const uint4*)(prow);
      aP.q[1] = *(const uint4*)(prow + 8);
    }
    // P @ V: contraction over 32 keys, N = 64 dims (4 tiles), V^T contiguous in s
    {
      const u16* vr = vbase + (size_t)(nIdx) * SEQ + j0 + 16 * g;
      Frag bv;
      bv.q[0] = *(const uint4*)(vr); bv.q[1] = *(const uint4*)(vr + 8);
      o0 = wmma_bf16(aP, bv, o0);
    }
    {
      const u16* vr = vbase + (size_t)(16 + nIdx) * SEQ + j0 + 16 * g;
      Frag bv;
      bv.q[0] = *(const uint4*)(vr); bv.q[1] = *(const uint4*)(vr + 8);
      o1 = wmma_bf16(aP, bv, o1);
    }
    {
      const u16* vr = vbase + (size_t)(32 + nIdx) * SEQ + j0 + 16 * g;
      Frag bv;
      bv.q[0] = *(const uint4*)(vr); bv.q[1] = *(const uint4*)(vr + 8);
      o2 = wmma_bf16(aP, bv, o2);
    }
    {
      const u16* vr = vbase + (size_t)(48 + nIdx) * SEQ + j0 + 16 * g;
      Frag bv;
      bv.q[0] = *(const uint4*)(vr); bv.q[1] = *(const uint4*)(vr + 8);
      o3 = wmma_bf16(aP, bv, o3);
    }
  }
  // normalize and store z (bf16, [b*S+s][E] with head at col h*64)
#pragma unroll
  for (int i = 0; i < 8; i++) {
    float inv = 1.0f / srow[i];
    size_t zr = ((size_t)(bS + q0 + i + 8 * g)) * EMBED + h * HDIM + nIdx;
    Z[zr]      = f2b(o0[i] * inv);
    Z[zr + 16] = f2b(o1[i] * inv);
    Z[zr + 32] = f2b(o2[i] * inv);
    Z[zr + 48] = f2b(o3[i] * inv);
  }
}

// ---------------------------------------------------------------------------
extern "C" void kernel_launch(void* const* d_in, const int* in_sizes, int n_in,
                              void* d_out, int out_size, void* d_ws, size_t ws_size,
                              hipStream_t stream) {
  (void)in_sizes; (void)n_in; (void)out_size; (void)ws_size;
  const float* x   = (const float*)d_in[0];
  const float* Wq  = (const float*)d_in[1];
  const float* bq  = (const float*)d_in[2];
  const float* Wk  = (const float*)d_in[3];
  const float* bk  = (const float*)d_in[4];
  const float* Wv  = (const float*)d_in[5];
  const float* bv  = (const float*)d_in[6];
  const float* Wo  = (const float*)d_in[7];
  const float* W1  = (const float*)d_in[8];
  const float* b1  = (const float*)d_in[9];
  const float* W2  = (const float*)d_in[10];
  const float* b2  = (const float*)d_in[11];
  const float* g1  = (const float*)d_in[12];
  const float* be1 = (const float*)d_in[13];
  const float* g2  = (const float*)d_in[14];
  const float* be2 = (const float*)d_in[15];
  float* out = (float*)d_out;

  char* ws = (char*)d_ws;
  size_t off = 0;
  auto alloc = [&](size_t bytes) -> char* {
    char* p = ws + off;
    off += (bytes + 255) & ~(size_t)255;
    return p;
  };
  const size_t EE = (size_t)EMBED * EMBED;
  const size_t EF = (size_t)EMBED * FFDIM;
  u16*   Wqb  = (u16*)alloc(EE * 2);
  u16*   Wkb  = (u16*)alloc(EE * 2);
  u16*   Wvb  = (u16*)alloc(EE * 2);
  u16*   Wob  = (u16*)alloc(EE * 2);
  u16*   W1b  = (u16*)alloc(EF * 2);
  u16*   W2b  = (u16*)alloc(EF * 2);
  u16*   xnb  = (u16*)alloc((size_t)MROWS * EMBED * 2);
  u16*   qb   = (u16*)alloc((size_t)MROWS * EMBED * 2);
  u16*   kb   = (u16*)alloc((size_t)MROWS * EMBED * 2);
  u16*   vTb  = (u16*)alloc((size_t)MROWS * EMBED * 2);
  u16*   zb   = (u16*)alloc((size_t)MROWS * EMBED * 2);
  float* hbuf = (float*)alloc((size_t)MROWS * EMBED * 4);
  u16*   mnb  = (u16*)alloc((size_t)MROWS * EMBED * 2);
  u16*   gbuf = (u16*)alloc((size_t)MROWS * FFDIM * 2);

  // 1) weights -> bf16
  cvt_bf16_k<<<(int)((EE + 255) / 256), 256, 0, stream>>>(Wq, Wqb, (int)EE);
  cvt_bf16_k<<<(int)((EE + 255) / 256), 256, 0, stream>>>(Wk, Wkb, (int)EE);
  cvt_bf16_k<<<(int)((EE + 255) / 256), 256, 0, stream>>>(Wv, Wvb, (int)EE);
  cvt_bf16_k<<<(int)((EE + 255) / 256), 256, 0, stream>>>(Wo, Wob, (int)EE);
  cvt_bf16_k<<<(int)((EF + 255) / 256), 256, 0, stream>>>(W1, W1b, (int)EF);
  cvt_bf16_k<<<(int)((EF + 255) / 256), 256, 0, stream>>>(W2, W2b, (int)EF);

  // 2) LN1(x) -> bf16
  layernorm_bf16_k<<<MROWS, 256, 0, stream>>>(x, g1, be1, xnb);

  // 3) Q/K/V projections (4096x1024x1024)
  dim3 gE(EMBED / 128, MROWS / 128);
  gemm_bf16_k<EP_BIAS | EP_B16><<<gE, 256, 0, stream>>>(
      xnb, Wqb, bq, nullptr, nullptr, qb, MROWS, EMBED, EMBED);
  gemm_bf16_k<EP_BIAS | EP_B16><<<gE, 256, 0, stream>>>(
      xnb, Wkb, bk, nullptr, nullptr, kb, MROWS, EMBED, EMBED);
  gemm_bf16_k<EP_BIAS | EP_VT><<<gE, 256, 0, stream>>>(
      xnb, Wvb, bv, nullptr, nullptr, vTb, MROWS, EMBED, EMBED);

  // 4) causal flash attention
  flash_attn_k<<<dim3(SEQ / 16, HEADS, BATCH), 32, 0, stream>>>(qb, kb, vTb, zb);

  // 5) h = x + z @ Wo
  gemm_bf16_k<EP_RES | EP_F32><<<gE, 256, 0, stream>>>(
      zb, Wob, nullptr, x, hbuf, nullptr, MROWS, EMBED, EMBED);

  // 6) LN2(h) -> bf16
  layernorm_bf16_k<<<MROWS, 256, 0, stream>>>(hbuf, g2, be2, mnb);

  // 7) g = gelu(mn @ W1 + b1)  (4096x4096x1024)
  dim3 gF(FFDIM / 128, MROWS / 128);
  gemm_bf16_k<EP_BIAS | EP_GELU | EP_B16><<<gF, 256, 0, stream>>>(
      mnb, W1b, b1, nullptr, nullptr, gbuf, MROWS, FFDIM, EMBED);

  // 8) out = h + g @ W2 + b2   (4096x1024x4096)
  gemm_bf16_k<EP_BIAS | EP_RES | EP_F32><<<gE, 256, 0, stream>>>(
      gbuf, W2b, b2, hbuf, out, nullptr, MROWS, EMBED, FFDIM);
}
